// CharRNN_83545703842005
// MI455X (gfx1250) — compile-verified, matrix operand order checked by reading, not run on hardware
//
#include <hip/hip_runtime.h>
#include <hip/hip_fp16.h>

typedef __attribute__((ext_vector_type(16))) _Float16 v16h;
typedef __attribute__((ext_vector_type(8)))  float    v8f;

#define SEQ 128
#define HID 64
#define G4  256     // 4*HID gates
#define NL  128
#define VOC 35

__device__ __forceinline__ float fast_rcp(float x) {
#if __has_builtin(__builtin_amdgcn_rcpf)
  return __builtin_amdgcn_rcpf(x);      // v_rcp_f32 (1 TRANS op)
#else
  return 1.0f / x;
#endif
}

__device__ __forceinline__ float sigmoidf_(float x) {
  return fast_rcp(1.0f + __expf(-x));   // v_exp + v_rcp, no div sequence
}

__device__ __forceinline__ float tanhf_(float x) {
#if __has_builtin(__builtin_amdgcn_tanhf)
  return __builtin_amdgcn_tanhf(x);     // gfx1250 v_tanh_f32
#elif __has_builtin(__builtin_amdgcn_tanh_f32)
  return __builtin_amdgcn_tanh_f32(x);
#else
  // tanh(x) = 1 - 2/(exp(2x)+1); exp saturates cleanly for |x| large
  const float e = __expf(2.0f * x);
  return 1.0f - 2.0f * fast_rcp(e + 1.0f);
#endif
}

__global__ __launch_bounds__(256, 1) void charrnn_lstm_kernel(
    const int*   __restrict__ x,
    const float* __restrict__ embed,
    const float* __restrict__ Wih0,
    const float* __restrict__ Whh0,
    const float* __restrict__ bih0,
    const float* __restrict__ bhh0,
    const float* __restrict__ Wih,
    const float* __restrict__ Whh,
    const float* __restrict__ bih,
    const float* __restrict__ bhh,
    const float* __restrict__ Wfc,
    const float* __restrict__ bfc,
    float* __restrict__ out,
    float* __restrict__ ws)
{
  // LDS: ~52 KB total
  __shared__ _Float16 ysA[SEQ * HID];     // 16 KB  layer I/O double buffer (f16 for WMMA A)
  __shared__ _Float16 ysB[SEQ * HID];     // 16 KB
  __shared__ float hbuf[HID];
  __shared__ float cbuf[HID];
  __shared__ float gates[G4];
  __shared__ float xeb[SEQ];
  __shared__ float colmax[VOC];
  __shared__ float colsum[VOC];
  __shared__ float elog[NL * VOC];        // 17.5 KB logits for the head

  float* __restrict__ Gx     = ws;             // SEQ*G4 f32 (128 KB): input-projection gates
  float* __restrict__ hidden = ws + SEQ * G4;  // NL*HID f32 (32 KB): final h per layer

  const int tid  = threadIdx.x;     // 0..255 : one gate per thread in phase B
  const int wave = tid >> 5;        // 0..7   : one M-tile per wave in phase A
  const int lane = tid & 31;
  const int lm   = lane & 15;       // m (A) / n (B,D) within 16x16 tile
  const int lh   = lane >> 4;       // half-wave selector (K/M split per ISA layout)

  // xe[t] = embed[x[t]]  (EMBED == 1)
  if (tid < SEQ) xeb[tid] = embed[x[tid]];
  __syncthreads();

  for (int layer = 0; layer < NL; ++layer) {
    const _Float16* ysin  = (layer & 1) ? ysA : ysB;
    _Float16*       ysout = (layer & 1) ? ysB : ysA;

    const float* Wl  = Wih;   // overwritten below for layer>0; unused for layer 0
    const float* Whl;
    float bsum, wx0 = 0.0f;
    if (layer == 0) {
      Whl  = Whh0;
      bsum = bih0[tid] + bhh0[tid];
      wx0  = Wih0[tid];                       // Wih0 is (256,1)
    } else {
      Wl   = Wih + (size_t)(layer - 1) * G4 * HID;
      Whl  = Whh + (size_t)(layer - 1) * G4 * HID;
      bsum = bih[(layer - 1) * G4 + tid] + bhh[(layer - 1) * G4 + tid];
    }

    // Hoist this thread's recurrent weight row into registers (64 VGPRs).
    float wr[HID];
    #pragma unroll
    for (int k = 0; k < HID; ++k) wr[k] = Whl[tid * HID + k];

    // ---------- Phase A: Gx[t][g] = sum_k ysin[t][k] * Wih[g][k]  (WMMA) ----------
    if (layer > 0) {
      const int mt = wave;                    // rows t = 16*mt .. 16*mt+15
      // A fragments (16x32 f16, ISA 7.12.2 layout), two K-slabs: 0..31 and 32..63
      v16h a0, a1;
      #pragma unroll
      for (int j = 0; j < 8; ++j) {
        const int k0  = (j < 4) ? (2 * j + 8 * lh) : (16 + 2 * (j - 4) + 8 * lh);
        const int row = (16 * mt + lm) * HID;
        a0[2 * j]     = ysin[row + k0];
        a0[2 * j + 1] = ysin[row + k0 + 1];
        a1[2 * j]     = ysin[row + 32 + k0];
        a1[2 * j + 1] = ysin[row + 32 + k0 + 1];
      }
      for (int nt = 0; nt < 16; ++nt) {       // 16 N-tiles of gates
        v16h b0, b1;
        const int col = (16 * nt + lm) * HID; // Wih row = gate index (B col)
        #pragma unroll
        for (int j = 0; j < 8; ++j) {
          const int k0 = (j < 4) ? (2 * j + 8 * lh) : (16 + 2 * (j - 4) + 8 * lh);
          b0[2 * j]     = (_Float16)Wl[col + k0];
          b0[2 * j + 1] = (_Float16)Wl[col + k0 + 1];
          b1[2 * j]     = (_Float16)Wl[col + 32 + k0];
          b1[2 * j + 1] = (_Float16)Wl[col + 32 + k0 + 1];
        }
        v8f c = {};
        c = __builtin_amdgcn_wmma_f32_16x16x32_f16(false, a0, false, b0,
                                                   (short)0, c, false, false);
        c = __builtin_amdgcn_wmma_f32_16x16x32_f16(false, a1, false, b1,
                                                   (short)0, c, false, false);
        #pragma unroll
        for (int r = 0; r < 8; ++r) {         // D: row m = r + 8*lh, col n = lm
          const int t = 16 * mt + r + 8 * lh;
          Gx[t * G4 + 16 * nt + lm] = c[r];
        }
      }
    }
    __threadfence_block();
    __syncthreads();

    // ---------- Phase B: sequential recurrence over t ----------
    if (tid < HID) { hbuf[tid] = 0.0f; cbuf[tid] = 0.0f; }
    __syncthreads();

    // Overlap the sequential sweep with pulling next layer's weights toward
    // the WGP: 256 threads x 256B stride covers the full 64KB of each matrix.
    if (layer + 1 < NL) {
      const size_t nb = (size_t)layer * G4 * HID;   // (layer+1) uses index layer
      __builtin_prefetch(Wih + nb + (size_t)tid * 64, 0, 1);
      __builtin_prefetch(Whh + nb + (size_t)tid * 64, 0, 1);
    }

    for (int t = 0; t < SEQ; ++t) {
      float acc = bsum;
      acc += (layer == 0) ? (wx0 * xeb[t]) : Gx[t * G4 + tid];
      #pragma unroll 8
      for (int k = 0; k < HID; ++k) acc += wr[k] * hbuf[k];
      gates[tid] = acc;
      __syncthreads();
      if (tid < HID) {                        // gate order i,f,g,o (PyTorch)
        const float ig = sigmoidf_(gates[tid]);
        const float fg = sigmoidf_(gates[HID + tid]);
        const float gg = tanhf_(gates[2 * HID + tid]);
        const float og = sigmoidf_(gates[3 * HID + tid]);
        const float cn = fg * cbuf[tid] + ig * gg;
        const float hn = og * tanhf_(cn);
        cbuf[tid] = cn;
        hbuf[tid] = hn;
        ysout[t * HID + tid] = (_Float16)hn;
      }
      __syncthreads();
    }

    if (tid < HID) hidden[layer * HID + tid] = hbuf[tid];
    __syncthreads();
  }

  // ---------- Head: logits, softmax across layers (axis 0), argmax over vocab ----------
  if (tid < NL) {
    float hl[HID];
    #pragma unroll
    for (int k = 0; k < HID; ++k) hl[k] = hidden[tid * HID + k];
    for (int v = 0; v < VOC; ++v) {
      float lg = bfc[v];
      #pragma unroll 8
      for (int k = 0; k < HID; ++k) lg += hl[k] * Wfc[v * HID + k];
      elog[tid * VOC + v] = lg;
    }
  }
  __syncthreads();
  if (tid < VOC) {                            // per-vocab column stats over 128 layers
    float mx = -3.4e38f;
    for (int l = 0; l < NL; ++l) mx = fmaxf(mx, elog[l * VOC + tid]);
    float s = 0.0f;
    for (int l = 0; l < NL; ++l) s += __expf(elog[l * VOC + tid] - mx);
    colmax[tid] = mx;
    colsum[tid] = s;
  }
  __syncthreads();
  if (tid < NL) {                             // argmax over vocab per layer row
    float best = -3.4e38f;
    int bi = 0;
    for (int v = 0; v < VOC; ++v) {
      const float p = __expf(elog[tid * VOC + v] - colmax[v]) / colsum[v];
      if (p > best) { best = p; bi = v; }     // strict > keeps first max (jnp.argmax)
    }
    out[tid] = (float)bi;
  }
}

extern "C" void kernel_launch(void* const* d_in, const int* in_sizes, int n_in,
                              void* d_out, int out_size, void* d_ws, size_t ws_size,
                              hipStream_t stream) {
  const int*   x     = (const int*)d_in[0];
  const float* embed = (const float*)d_in[1];
  const float* Wih0  = (const float*)d_in[2];
  const float* Whh0  = (const float*)d_in[3];
  const float* bih0  = (const float*)d_in[4];
  const float* bhh0  = (const float*)d_in[5];
  const float* Wih   = (const float*)d_in[6];
  const float* Whh   = (const float*)d_in[7];
  const float* bih   = (const float*)d_in[8];
  const float* bhh   = (const float*)d_in[9];
  const float* Wfc   = (const float*)d_in[10];
  const float* bfc   = (const float*)d_in[11];
  float* out = (float*)d_out;
  float* ws  = (float*)d_ws;   // needs SEQ*G4 + NL*HID floats = 160 KB

  charrnn_lstm_kernel<<<1, 256, 0, stream>>>(x, embed, Wih0, Whh0, bih0, bhh0,
                                             Wih, Whh, bih, bhh, Wfc, bfc, out, ws);
}
